// TwoStageBiRecurrentBackbone_19619410608641
// MI455X (gfx1250) — compile-verified, matrix operand order checked by reading, not run on hardware
//
#include <hip/hip_runtime.h>

#define T_   12
#define C_   64
#define H_   112
#define W_   112
#define HW_  (H_*W_)
#define PH_  114          // padded spatial (1-pixel zero border)
#define LRELU 0.1f

typedef __attribute__((ext_vector_type(8)))  float          v8f;
typedef __attribute__((ext_vector_type(16))) __bf16         v16bf;
typedef __attribute__((ext_vector_type(8)))  unsigned short u16x8;
typedef __attribute__((ext_vector_type(16))) unsigned short u16x16;

union FragU { u16x16 u; v16bf b; };

static __device__ __forceinline__ unsigned short f2bf(float f) {
  union { float f; unsigned u; } v; v.f = f;
  unsigned r = v.u + 0x7FFFu + ((v.u >> 16) & 1u);   // round-to-nearest-even bf16
  return (unsigned short)(r >> 16);
}

// ---------------------------------------------------------------------------
// Weight packing: OIHW f32 -> bf16 in exact WMMA B-fragment lane order.
// Read layout in conv: wp[((tap*KC + kc)*4 + nt)*512 + lane*16 + j]
//   co = nt*16 + (lane&15);  k = (lane>>4)*8 + j + (j>=8 ? 8 : 0); ci = kc*32+k
// ---------------------------------------------------------------------------
template<int CIN>
__global__ void pack_weights(const float* __restrict__ Wsrc,
                             unsigned short* __restrict__ dst) {
  constexpr int KC = CIN / 32;
  int id = blockIdx.x * blockDim.x + threadIdx.x;
  const int total = 9 * KC * 4 * 512;
  if (id >= total) return;
  int j    = id & 15;
  int lane = (id >> 4) & 31;
  int nt   = (id >> 9) & 3;
  int rest = id >> 11;
  int kc   = rest % KC;
  int tap  = rest / KC;
  int co = nt * 16 + (lane & 15);
  int k  = ((lane >> 4) * 8) + j + ((j >= 8) ? 8 : 0);
  int ci = kc * 32 + k;
  dst[id] = f2bf(Wsrc[((size_t)(co * CIN + ci)) * 9 + tap]);
}

// ---------------------------------------------------------------------------
// Bilinear warp setup (matches reference flow_warp: per-corner validity mask,
// clamped gather index, weight folded with validity)
// ---------------------------------------------------------------------------
struct WarpS { int i00, i01, i10, i11; float w00, w01, w10, w11; };

static __device__ __forceinline__ int clampi(int x, int lo, int hi) {
  return x < lo ? lo : (x > hi ? hi : x);
}

static __device__ __forceinline__ WarpS warp_setup(float vx, float vy) {
  float x0f = floorf(vx), y0f = floorf(vy);
  float wx = vx - x0f, wy = vy - y0f;
  int x0 = (int)x0f, y0 = (int)y0f;
  int x1 = x0 + 1,  y1 = y0 + 1;
  WarpS s;
  s.i00 = clampi(y0,0,H_-1)*W_ + clampi(x0,0,W_-1);
  s.i01 = clampi(y0,0,H_-1)*W_ + clampi(x1,0,W_-1);
  s.i10 = clampi(y1,0,H_-1)*W_ + clampi(x0,0,W_-1);
  s.i11 = clampi(y1,0,H_-1)*W_ + clampi(x1,0,W_-1);
  float v00 = (x0>=0 && x0<W_ && y0>=0 && y0<H_) ? 1.f : 0.f;
  float v01 = (x1>=0 && x1<W_ && y0>=0 && y0<H_) ? 1.f : 0.f;
  float v10 = (x0>=0 && x0<W_ && y1>=0 && y1<H_) ? 1.f : 0.f;
  float v11 = (x1>=0 && x1<W_ && y1>=0 && y1<H_) ? 1.f : 0.f;
  s.w00 = (1.f-wx)*(1.f-wy)*v00;
  s.w01 = wx*(1.f-wy)*v01;
  s.w10 = (1.f-wx)*wy*v10;
  s.w11 = wx*wy*v11;
  return s;
}

// ---------------------------------------------------------------------------
// f2_tot = f1 + flow_warp(f2, f1)   (2-channel flow composition)
// ---------------------------------------------------------------------------
__global__ void flow2_kernel(const float* __restrict__ flows,
                             float* __restrict__ f2t, int tf1, int tf2) {
  int idx = blockIdx.x * blockDim.x + threadIdx.x;
  if (idx >= 2 * HW_) return;
  int n = idx / HW_, hw = idx % HW_;
  int h = hw / W_,  w  = hw % W_;
  const float* f1 = flows + ((size_t)(n*T_ + tf1) * 2) * HW_;
  const float* f2 = flows + ((size_t)(n*T_ + tf2) * 2) * HW_;
  float fx = f1[hw], fy = f1[HW_ + hw];
  WarpS s = warp_setup((float)w + fx, (float)h + fy);
  float wx2 = s.w00*f2[s.i00] + s.w01*f2[s.i01] + s.w10*f2[s.i10] + s.w11*f2[s.i11];
  const float* f2y = f2 + HW_;
  float wy2 = s.w00*f2y[s.i00] + s.w01*f2y[s.i01] + s.w10*f2y[s.i10] + s.w11*f2y[s.i11];
  f2t[(size_t)(n*2+0)*HW_ + hw] = fx + wx2;
  f2t[(size_t)(n*2+1)*HW_ + hw] = fy + wy2;
}

// ---------------------------------------------------------------------------
// Build aggr = [curr, warp(p1,f1), warp(p2,f2tot)] as bf16 padded NHWC
// aggr[n][h+1][w+1][c] (c in 0..191), border stays zero (memset once).
// ---------------------------------------------------------------------------
__global__ void aggr_kernel(const float* __restrict__ feats,
                            const float* __restrict__ flows,
                            const float* __restrict__ f2t,
                            unsigned short* __restrict__ aggr,
                            int t, int t1, int t2, int tf1,
                            int use1, int use2) {
  int idx = blockIdx.x * blockDim.x + threadIdx.x;
  if (idx >= 2 * HW_) return;
  int n = idx / HW_, hw = idx % HW_;
  int h = hw / W_,  w  = hw % W_;
  const float* cur = feats + ((size_t)(n*T_ + t )*C_) * HW_ + hw;
  const float* p1  = feats + ((size_t)(n*T_ + t1)*C_) * HW_;
  const float* p2  = feats + ((size_t)(n*T_ + t2)*C_) * HW_;
  WarpS s1 = {}; WarpS s2 = {};
  if (use1) {
    const float* f1 = flows + ((size_t)(n*T_ + tf1) * 2) * HW_;
    s1 = warp_setup((float)w + f1[hw], (float)h + f1[HW_ + hw]);
  }
  if (use2) {
    float fx = f2t[(size_t)(n*2+0)*HW_ + hw];
    float fy = f2t[(size_t)(n*2+1)*HW_ + hw];
    s2 = warp_setup((float)w + fx, (float)h + fy);
  }
  unsigned short* ab = aggr + ((size_t)(n*PH_ + (h+1))*PH_ + (w+1)) * (3*C_);
  for (int c = 0; c < C_; ++c) {
    float cv = cur[(size_t)c * HW_];
    const float* pc1 = p1 + (size_t)c * HW_;
    const float* pc2 = p2 + (size_t)c * HW_;
    float a1 = use1 ? (s1.w00*pc1[s1.i00] + s1.w01*pc1[s1.i01] +
                       s1.w10*pc1[s1.i10] + s1.w11*pc1[s1.i11]) : cv;
    float a2 = use2 ? (s2.w00*pc2[s2.i00] + s2.w01*pc2[s2.i01] +
                       s2.w10*pc2[s2.i10] + s2.w11*pc2[s2.i11]) : cv;
    ab[c]        = f2bf(cv);
    ab[C_ + c]   = f2bf(a1);
    ab[2*C_ + c] = f2bf(a2);
  }
}

// ---------------------------------------------------------------------------
// Implicit-GEMM 3x3 conv via v_wmma_f32_16x16x32_bf16 with DOUBLE-BUFFERED
// async global->LDS staging (GLOBAL_LOAD_ASYNC_TO_LDS_B128 / ASYNCcnt).
// One block (7 waves, 224 threads) = one output row (n,h): wave wv owns the
// 16-wide w-tile, all 64 output channels (4 N-tiles).
// Pipeline: issue row dh+1 into LDS buf[(dh+1)&1] right after the barrier,
// compute taps of row dh from buf[dh&1]; per-wave s_wait_asynccnt 0 + barrier
// fences both DMA completion and buffer reuse.
// LDS column stride padded to CIN+8 halves -> near conflict-free ds_load_b128.
// MODE 0: +bias, leaky-ReLU, store bf16 padded NHWC (conv1 -> h1)
// MODE 1: +bias + residual(curr), store f32 into feats[n][t] (conv2)
// ---------------------------------------------------------------------------
template<int CIN, int MODE>
__global__ __launch_bounds__(224) void conv_wmma(
    const unsigned short* __restrict__ in,   // padded NHWC bf16 [n][114][114][CIN]
    const unsigned short* __restrict__ wp,   // packed weights
    const float* __restrict__ bias,
    float* __restrict__ feats,               // MODE 1 in/out
    unsigned short* __restrict__ h1out,      // MODE 0 out
    int t) {
  constexpr int KC   = CIN / 32;
  constexpr int CSTR = CIN + 8;              // padded per-column halves in LDS
  constexpr int C8   = CIN / 8;              // 16B chunks per column
  constexpr unsigned BUFH = PH_ * CSTR;      // halves per buffer
  __shared__ unsigned short sA[2 * BUFH];    // double buffer (starts at LDS 0)

  int tid  = threadIdx.x;
  int lane = tid & 31;
  int wv   = tid >> 5;                       // 0..6
  int n    = blockIdx.x / H_;
  int h    = blockIdx.x % H_;
  int wb   = wv * 16;
  int lm = lane & 15, hi = lane >> 4;
  const unsigned short* inN = in + (size_t)n * PH_ * PH_ * CIN;

  // async-issue one padded input row into LDS buffer bufi (per-lane b128 DMA)
  auto issue_row = [&](int dh, int bufi) {
    unsigned long long g =
        (unsigned long long)(inN + ((size_t)(h + dh) * PH_) * CIN);
    unsigned bbyte = (unsigned)bufi * (BUFH * 2u);
    for (int idx = tid; idx < PH_ * C8; idx += 224) {
      int col = idx / C8, c8 = idx % C8;
      unsigned voff = (unsigned)((col * CIN + c8 * 8) * 2);
      unsigned ldsa = bbyte + (unsigned)((col * CSTR + c8 * 8) * 2);
      asm volatile("global_load_async_to_lds_b128 %0, %1, %2"
                   :: "v"(ldsa), "v"(voff), "s"(g) : "memory");
    }
  };

  v8f acc[4];
  #pragma unroll
  for (int i = 0; i < 4; ++i) acc[i] = 0.f;

  issue_row(0, 0);                           // prologue

  for (int dh = 0; dh < 3; ++dh) {
    asm volatile("s_wait_asynccnt 0x0" ::: "memory");  // my DMAs landed
    __syncthreads();                                   // everyone's landed
    if (dh < 2) issue_row(dh + 1, (dh + 1) & 1);       // prefetch next row
    const unsigned short* buf = sA + (unsigned)(dh & 1) * BUFH;

    #pragma unroll
    for (int dw = 0; dw < 3; ++dw) {
      int tap = dh * 3 + dw;
      const unsigned short* abase = buf + (wb + lm + dw) * CSTR + hi * 8;
      const unsigned short* bbase = wp + (size_t)tap * KC * 4 * 512 + lane * 16;
      // prefetch next tap's weight block into cache (global_prefetch_b8)
      if (tap < 8)
        __builtin_prefetch((const void*)(bbase + (size_t)KC * 4 * 512), 0, 1);
      #pragma unroll
      for (int kc = 0; kc < KC; ++kc) {
        u16x8 a0 = *(const u16x8*)(abase + kc * 32);        // ds_load_b128
        u16x8 a1 = *(const u16x8*)(abase + kc * 32 + 16);   // ds_load_b128
        FragU fa;
        fa.u = __builtin_shufflevector(a0, a1,
                 0,1,2,3,4,5,6,7,8,9,10,11,12,13,14,15);
        const unsigned short* bk = bbase + (size_t)kc * 4 * 512;
        #pragma unroll
        for (int nt = 0; nt < 4; ++nt) {
          FragU fb; fb.u = *(const u16x16*)(bk + nt * 512);
          acc[nt] = __builtin_amdgcn_wmma_f32_16x16x32_bf16(
              false, fa.b, false, fb.b, (short)0, acc[nt], false, false);
        }
      }
    }
  }

  if (MODE == 0) {
    unsigned short* ob =
        h1out + ((size_t)(n * PH_ + (h + 1)) * PH_ + (wb + 1)) * C_;
    #pragma unroll
    for (int nt = 0; nt < 4; ++nt) {
      int co = nt * 16 + lm;
      float b = bias[co];
      #pragma unroll
      for (int r = 0; r < 8; ++r) {
        int wl = hi * 8 + r;                 // 0..15 within tile (D layout)
        float v = acc[nt][r] + b;
        v = v > 0.f ? v : LRELU * v;
        ob[(size_t)wl * C_ + co] = f2bf(v);
      }
    }
  } else {
    float* yb = feats + ((size_t)(n * T_ + t) * C_) * HW_;
    #pragma unroll
    for (int nt = 0; nt < 4; ++nt) {
      int co = nt * 16 + lm;
      float b = bias[co];
      size_t base = (size_t)co * HW_ + (size_t)h * W_ + wb + hi * 8;
      #pragma unroll
      for (int r = 0; r < 8; ++r) {
        yb[base + r] = acc[nt][r] + b + yb[base + r];   // + residual curr
      }
    }
  }
}

// ---------------------------------------------------------------------------
extern "C" void kernel_launch(void* const* d_in, const int* in_sizes, int n_in,
                              void* d_out, int out_size, void* d_ws, size_t ws_size,
                              hipStream_t stream) {
  const float* feats_in = (const float*)d_in[0];
  const float* ffl = (const float*)d_in[1];
  const float* bfl = (const float*)d_in[2];
  const float* W1  = (const float*)d_in[3];
  const float* b1  = (const float*)d_in[4];
  const float* W2  = (const float*)d_in[5];
  const float* b2  = (const float*)d_in[6];
  float* feats = (float*)d_out;              // recurrent state lives in d_out

  char* ws = (char*)d_ws;
  unsigned short* wp1  = (unsigned short*)(ws);                 // 221184 B
  unsigned short* wp2  = (unsigned short*)(ws + 221184);        //  73728 B
  unsigned short* aggr = (unsigned short*)(ws + 294912);        // 9983232 B
  unsigned short* h1   = (unsigned short*)(ws + 10278144);      // 3327744 B
  float*          f2t  = (float*)        (ws + 13605888);       //  200704 B

  hipMemcpyAsync(feats, feats_in, (size_t)2*T_*C_*HW_*sizeof(float),
                 hipMemcpyDeviceToDevice, stream);
  hipMemsetAsync(aggr, 0, (size_t)2*PH_*PH_*3*C_*2, stream);
  hipMemsetAsync(h1,   0, (size_t)2*PH_*PH_*C_*2,   stream);

  pack_weights<192><<<(9*6*4*512 + 255)/256, 256, 0, stream>>>(W1, wp1);
  pack_weights< 64><<<(9*2*4*512 + 255)/256, 256, 0, stream>>>(W2, wp2);

  const int npix = 2 * HW_;
  dim3 gp((npix + 255) / 256), bp(256);
  dim3 gc(2 * H_), bc(224);                  // one block per output row

  for (int pass = 0; pass < 2; ++pass) {
    const float* flows = (pass == 0) ? ffl : bfl;
    for (int i = 0; i < T_; ++i) {
      int t, t1, t2, tf1, tf2;
      if (pass == 0) { t = i;        t1 = i - 1; t2 = i - 2; tf1 = i - 1; tf2 = i - 2; }
      else           { t = T_-1 - i; t1 = t + 1; t2 = t + 2; tf1 = t + 1; tf2 = t + 2; }
      int use1 = (i > 0), use2 = (i > 1);
      int t1c  = use1 ? t1  : t;
      int t2c  = use2 ? t2  : t;
      int tf1c = use1 ? tf1 : 0;
      int tf2c = use2 ? tf2 : 0;
      if (use2)
        flow2_kernel<<<gp, bp, 0, stream>>>(flows, f2t, tf1c, tf2c);
      aggr_kernel<<<gp, bp, 0, stream>>>(feats, flows, f2t, aggr,
                                         t, t1c, t2c, tf1c, use1, use2);
      conv_wmma<192, 0><<<gc, bc, 0, stream>>>(aggr, wp1, b1, nullptr, h1, t);
      conv_wmma< 64, 1><<<gc, bc, 0, stream>>>(h1,   wp2, b2, feats, nullptr, t);
    }
  }
}
